// LSTMMeasurementPredictor_53678501265897
// MI455X (gfx1250) — matrix operand order repaired and checked by reading, not compiled, not run on hardware
//
#include <hip/hip_runtime.h>
#include <hip/hip_bf16.h>

// ---------------- model dimensions ----------------
#define BQ     8192          // batch
#define HHID   256           // hidden size
#define GGATE  1024          // 4*H gate width
#define TSTEPS 16            // time steps
#define BT     16            // batch rows per workgroup tile
#define NBTILE (BQ / BT)     // 512 tiles

// LDS padding strides (avoid bank conflicts on row-gathers)
#define GSTR 1028            // f32 gate rows
#define HSTR 264             // bf16 256-wide rows
#define XSTR 34              // bf16 32-wide rows

#define RECON_ELEMS (BQ * TSTEPS * 32)   // 4,194,304 f32, then bases (re/im interleaved)

typedef __attribute__((ext_vector_type(16))) __bf16 v16bf;
typedef __attribute__((ext_vector_type(8)))  float  v8f;

__device__ __forceinline__ float sigm(float x) { return 1.0f / (1.0f + __expf(-x)); }

__device__ __forceinline__ v8f wmma_bf16(v16bf a, v16bf b, v8f c) {
  // D(f32 16x16) = A(bf16 16x32) x B(bf16 32x16) + C
  return __builtin_amdgcn_wmma_f32_16x16x32_bf16(false, a, false, b, (short)0, c, false, false);
}

// Gather a 16x32 bf16 A fragment from an LDS row-major tile (row stride xstr).
// ISA layout: lane l -> row M = l%16; element i -> K = (i/8)*16 + (l>=16?8:0) + i%8 (+kbase).
__device__ __forceinline__ v16bf load_afrag(const __bf16* x, int lane, int xstr, int kbase) {
  const int m    = lane & 15;
  const int koff = kbase + ((lane >> 4) << 3);
  const __bf16* p = x + m * xstr + koff;
  v16bf a;
#pragma unroll
  for (int g = 0; g < 2; ++g)
#pragma unroll
    for (int w = 0; w < 8; ++w)
      a[g * 8 + w] = p[g * 16 + w];
  return a;
}

// Store a 16x32 bf16 tile (LDS, row stride xstr) into memory in A-fragment order:
// dst[lane*16 + i] = x[M(lane), K(lane,i)].  512 halves, 2 per thread of a 256-block.
__device__ __forceinline__ void store_xfrag(__bf16* dst, const __bf16* x, int xstr, int tid) {
#pragma unroll
  for (int r = 0; r < 2; ++r) {
    const int id = tid * 2 + r;          // 0..511
    const int i  = id & 15;
    const int ln = id >> 4;
    const int m  = ln & 15;
    const int k  = ((i >> 3) << 4) + (((ln >> 4) & 1) << 3) + (i & 7);
    dst[id] = x[m * xstr + k];
  }
}

// ---------------- weight pre-swizzle into WMMA B-fragment order ----------------
// dst layout: ((kc*NT + nt)*32 + lane)*16 + i ;  B[k,n] = W[n*K + k] (zero-pad k>=K).
// ISA B layout: lane l -> col N = l%16; element i -> K = kc*32 + (i/8)*16 + (l>=16?8:0) + i%8.
__global__ __launch_bounds__(256, 1)
void fill_bfrag(const float* __restrict__ W, __bf16* __restrict__ dst,
                int NT, int KC, int K) {
  const int id = blockIdx.x * blockDim.x + threadIdx.x;
  const int total = KC * NT * 512;
  if (id >= total) return;
  const int i    = id & 15;
  const int lane = (id >> 4) & 31;
  const int nt   = (id >> 9) % NT;
  const int kc   = id / (NT * 512);
  const int n = nt * 16 + (lane & 15);
  const int k = kc * 32 + ((i >> 3) << 4) + ((lane >> 4) & 1) * 8 + (i & 7);
  const float v = (k < K) ? W[n * K + k] : 0.0f;
  dst[id] = (__bf16)v;
}

// ---------------- per-row 2-qubit measurement algebra ----------------
// v[8]: (q, j, re/im). Builds trace-normalized bm (as 16 interleaved floats) and
// returns m = Re Tr((bm0 kron bm1) rho).
__device__ __forceinline__ float quantum_step(const float* v, const float* rho, float* bmri) {
#pragma unroll
  for (int q = 0; q < 2; ++q) {
    const float ar = v[q * 4 + 0], ai = v[q * 4 + 1];   // b0
    const float br = v[q * 4 + 2], bi = v[q * 4 + 3];   // b1
    const float n0 = ar * ar + ai * ai;
    const float n1 = br * br + bi * bi;
    const float inv = 1.0f / (n0 + n1);
    const float c01r = (ar * br + ai * bi) * inv;       // b0*conj(b1)/tr
    const float c01i = (ai * br - ar * bi) * inv;
    bmri[q * 8 + 0] = n0 * inv;  bmri[q * 8 + 1] = 0.0f;
    bmri[q * 8 + 2] = c01r;      bmri[q * 8 + 3] = c01i;
    bmri[q * 8 + 4] = c01r;      bmri[q * 8 + 5] = -c01i;
    bmri[q * 8 + 6] = n1 * inv;  bmri[q * 8 + 7] = 0.0f;
  }
  float mre = 0.0f;
#pragma unroll
  for (int i = 0; i < 2; ++i)
#pragma unroll
    for (int j = 0; j < 2; ++j) {
      const float x0r = bmri[(i * 2 + j) * 2 + 0];
      const float x0i = bmri[(i * 2 + j) * 2 + 1];
#pragma unroll
      for (int k = 0; k < 2; ++k)
#pragma unroll
        for (int l = 0; l < 2; ++l) {
          const float x1r = bmri[8 + (k * 2 + l) * 2 + 0];
          const float x1i = bmri[8 + (k * 2 + l) * 2 + 1];
          const float pr = x0r * x1r - x0i * x1i;
          const float pi = x0r * x1i + x0i * x1r;
          const int rrow = j * 2 + l, rcol = i * 2 + k;   // rho[j*2+l][i*2+k]
          mre += pr * rho[rrow * 4 + rcol] - pi * rho[16 + rrow * 4 + rcol];
        }
    }
  return mre;
}

// ---------------- Phase A: measurement-predictor LSTM cell (15 recurrent steps) -----
__global__ __launch_bounds__(256, 1)
void qst_phaseA(const float* __restrict__ meas, const float* __restrict__ basis_ri,
                const float* __restrict__ rho,  const float* __restrict__ h0,
                const float* __restrict__ c0,   const float* __restrict__ cell_b,
                const float* __restrict__ proj_W, const float* __restrict__ proj_b,
                const __bf16* __restrict__ fWih, const __bf16* __restrict__ fWhh,
                __bf16* __restrict__ seq_frag,   float* __restrict__ outF) {
  __shared__ float  g_lds[BT * GSTR];
  __shared__ __bf16 h_bf [BT * HSTR];
  __shared__ __bf16 x_bf [BT * XSTR];
  __shared__ float  rho_s[BT * 32];
  __shared__ float  v_s  [BT * 8];
  __shared__ float  pw_s [HHID * 8];     // proj_W transposed: pw_s[k*8+p]

  const int tile = blockIdx.x;
  const int b0   = tile * BT;
  const int tid  = threadIdx.x;
  const int lane = tid & 31;
  const int wave = tid >> 5;

  float bval[8];
#pragma unroll
  for (int nt = 0; nt < 8; ++nt)
    bval[nt] = cell_b[wave * 128 + nt * 16 + (lane & 15)];

  float c_reg[BT];
#pragma unroll
  for (int m = 0; m < BT; ++m) {
    c_reg[m] = c0[(size_t)(b0 + m) * HHID + tid];
    h_bf[m * HSTR + tid] = (__bf16)h0[(size_t)(b0 + m) * HHID + tid];
  }
  rho_s[tid]       = rho[(size_t)b0 * 32 + tid];
  rho_s[tid + 256] = rho[(size_t)b0 * 32 + 256 + tid];
  for (int idx = tid; idx < HHID * 8; idx += 256)
    pw_s[idx] = proj_W[(idx & 7) * HHID + (idx >> 3)];

  if (tid < BT) {                         // t = 0 input + outputs
    const int m = tid, b = b0 + m;
    x_bf[m * XSTR + 0] = (__bf16)meas[b];
    float* bout = outF + RECON_ELEMS + ((size_t)b * TSTEPS + 0) * 16;
#pragma unroll
    for (int q = 0; q < 16; ++q) {
      const float bb = basis_ri[(size_t)b * 16 + q];
      x_bf[m * XSTR + 1 + q] = (__bf16)bb;
      bout[q] = bb;
    }
#pragma unroll
    for (int q = 17; q < 32; ++q) x_bf[m * XSTR + q] = (__bf16)0.0f;
  }
  __syncthreads();
  store_xfrag(seq_frag + (size_t)tile * TSTEPS * 512, x_bf, XSTR, tid);

  for (int t = 1; t < TSTEPS; ++t) {
    __syncthreads();
    // ---- gate GEMM: g = x@Wih^T + h@Whh^T + b  (wave owns 8 N-tiles) ----
    v8f acc[8];
#pragma unroll
    for (int nt = 0; nt < 8; ++nt)
#pragma unroll
      for (int r = 0; r < 8; ++r) acc[nt][r] = bval[nt];
    {
      const v16bf a = load_afrag(x_bf, lane, XSTR, 0);
#pragma unroll
      for (int nt = 0; nt < 8; ++nt) {
        const v16bf b = *(const v16bf*)(fWih + ((size_t)(wave * 8 + nt) * 32 + lane) * 16);
        acc[nt] = wmma_bf16(a, b, acc[nt]);
      }
    }
    for (int kc = 0; kc < 8; ++kc) {
      if (kc < 7)
        __builtin_prefetch((const void*)(fWhh + ((size_t)((kc + 1) * 64 + wave * 8) * 32 + lane) * 16), 0, 1);
      const v16bf a = load_afrag(h_bf, lane, HSTR, kc * 32);
#pragma unroll
      for (int nt = 0; nt < 8; ++nt) {
        const v16bf b = *(const v16bf*)(fWhh + ((size_t)(kc * 64 + wave * 8 + nt) * 32 + lane) * 16);
        acc[nt] = wmma_bf16(a, b, acc[nt]);
      }
    }
    const int mb = (lane >> 4) << 3;      // C layout: row = r + 8*(lane>=16)
#pragma unroll
    for (int nt = 0; nt < 8; ++nt) {
      const int col = wave * 128 + nt * 16 + (lane & 15);
#pragma unroll
      for (int r = 0; r < 8; ++r) g_lds[(mb + r) * GSTR + col] = acc[nt][r];
    }
    __syncthreads();
    // ---- LSTM gates (thread tid owns hidden unit tid for all 16 rows) ----
#pragma unroll
    for (int m = 0; m < BT; ++m) {
      const float gi = g_lds[m * GSTR + tid];
      const float gf = g_lds[m * GSTR + 256 + tid];
      const float gg = g_lds[m * GSTR + 512 + tid];
      const float go = g_lds[m * GSTR + 768 + tid];
      const float cc = sigm(gf) * c_reg[m] + sigm(gi) * tanhf(gg);
      c_reg[m] = cc;
      h_bf[m * HSTR + tid] = (__bf16)(sigm(go) * tanhf(cc));
    }
    __syncthreads();
    // ---- v = h @ proj_W^T + proj_b : 128 dots of length 256 ----
    if (tid < 128) {
      const int m = tid >> 3, p = tid & 7;
      float d = proj_b[p];
      for (int k = 0; k < HHID; ++k)
        d += (float)h_bf[m * HSTR + k] * pw_s[k * 8 + p];
      v_s[m * 8 + p] = d;
    }
    __syncthreads();
    // ---- measurement algebra, next input, bases output ----
    if (tid < BT) {
      const int m = tid;
      float bmri[16];
      const float mre = quantum_step(&v_s[m * 8], &rho_s[m * 32], bmri);
      x_bf[m * XSTR + 0] = (__bf16)mre;
      float* bout = outF + RECON_ELEMS + ((size_t)(b0 + m) * TSTEPS + t) * 16;
#pragma unroll
      for (int q = 0; q < 16; ++q) {
        x_bf[m * XSTR + 1 + q] = (__bf16)bmri[q];
        bout[q] = bmri[q];
      }
    }
    __syncthreads();
    store_xfrag(seq_frag + ((size_t)tile * TSTEPS + t) * 512, x_bf, XSTR, tid);
  }
}

// ---------------- Phase B/C: projection LSTM layer (h=0,c=0 init, 16 steps) --------
__global__ __launch_bounds__(256, 1)
void qst_phaseB(const __bf16* __restrict__ xfrag, const __bf16* __restrict__ fWih,
                const __bf16* __restrict__ fWhh,  const __bf16* __restrict__ fWhr,
                const float* __restrict__ bias,
                __bf16* __restrict__ out_frag, float* __restrict__ out_f32) {
  __shared__ float  g_lds[BT * GSTR];
  __shared__ __bf16 s_bf [BT * HSTR];     // sigmoid(o)*tanh(c), 16x256
  __shared__ __bf16 hs_bf[BT * XSTR];     // projected h, 16x32
  __shared__ float  hsm  [BT * 32];

  const int tile = blockIdx.x;
  const int b0   = tile * BT;
  const int tid  = threadIdx.x;
  const int lane = tid & 31;
  const int wave = tid >> 5;

  float bval[8];
#pragma unroll
  for (int nt = 0; nt < 8; ++nt)
    bval[nt] = bias[wave * 128 + nt * 16 + (lane & 15)];

  float c_reg[BT];
#pragma unroll
  for (int m = 0; m < BT; ++m) c_reg[m] = 0.0f;
#pragma unroll
  for (int r = 0; r < 2; ++r) {
    const int idx = tid + r * 256;
    hs_bf[(idx >> 5) * XSTR + (idx & 31)] = (__bf16)0.0f;
  }
  __syncthreads();

  for (int t = 0; t < TSTEPS; ++t) {
    v8f acc[8];
#pragma unroll
    for (int nt = 0; nt < 8; ++nt)
#pragma unroll
      for (int r = 0; r < 8; ++r) acc[nt][r] = bval[nt];
    // input fragment is pre-swizzled in memory: single 32B vector load per lane
    const v16bf ax = *(const v16bf*)(xfrag + ((size_t)tile * TSTEPS + t) * 512 + lane * 16);
    const v16bf ah = load_afrag(hs_bf, lane, XSTR, 0);
#pragma unroll
    for (int nt = 0; nt < 8; ++nt) {
      const v16bf bih = *(const v16bf*)(fWih + ((size_t)(wave * 8 + nt) * 32 + lane) * 16);
      acc[nt] = wmma_bf16(ax, bih, acc[nt]);
      const v16bf bhh = *(const v16bf*)(fWhh + ((size_t)(wave * 8 + nt) * 32 + lane) * 16);
      acc[nt] = wmma_bf16(ah, bhh, acc[nt]);
    }
    const int mb = (lane >> 4) << 3;
#pragma unroll
    for (int nt = 0; nt < 8; ++nt) {
      const int col = wave * 128 + nt * 16 + (lane & 15);
#pragma unroll
      for (int r = 0; r < 8; ++r) g_lds[(mb + r) * GSTR + col] = acc[nt][r];
    }
    __syncthreads();
#pragma unroll
    for (int m = 0; m < BT; ++m) {
      const float gi = g_lds[m * GSTR + tid];
      const float gf = g_lds[m * GSTR + 256 + tid];
      const float gg = g_lds[m * GSTR + 512 + tid];
      const float go = g_lds[m * GSTR + 768 + tid];
      const float cc = sigm(gf) * c_reg[m] + sigm(gi) * tanhf(gg);
      c_reg[m] = cc;
      s_bf[m * HSTR + tid] = (__bf16)(sigm(go) * tanhf(cc));
    }
    __syncthreads();
    // ---- projection h = s @ Whr^T : M=16, N=32 (2 tiles on waves 0/1), K=256 ----
    if (wave < 2) {
      v8f a2;
#pragma unroll
      for (int r = 0; r < 8; ++r) a2[r] = 0.0f;
      for (int kc = 0; kc < 8; ++kc) {
        const v16bf a = load_afrag(s_bf, lane, HSTR, kc * 32);
        const v16bf b = *(const v16bf*)(fWhr + ((size_t)(kc * 2 + wave) * 32 + lane) * 16);
        a2 = wmma_bf16(a, b, a2);
      }
      const int col = wave * 16 + (lane & 15);
#pragma unroll
      for (int r = 0; r < 8; ++r) hsm[(mb + r) * 32 + col] = a2[r];
    }
    __syncthreads();
#pragma unroll
    for (int r2 = 0; r2 < 2; ++r2) {
      const int idx = tid + r2 * 256;
      hs_bf[(idx >> 5) * XSTR + (idx & 31)] = (__bf16)hsm[idx];
    }
    if (out_f32) {
#pragma unroll
      for (int r2 = 0; r2 < 2; ++r2) {
        const int idx = tid + r2 * 256;
        out_f32[((size_t)(b0 + (idx >> 5)) * TSTEPS + t) * 32 + (idx & 31)] = hsm[idx];
      }
    }
    __syncthreads();
    if (out_frag)
      store_xfrag(out_frag + ((size_t)tile * TSTEPS + t) * 512, hs_bf, XSTR, tid);
    __syncthreads();
  }
}

// ---------------- host launcher ----------------
extern "C" void kernel_launch(void* const* d_in, const int* in_sizes, int n_in,
                              void* d_out, int out_size, void* d_ws, size_t ws_size,
                              hipStream_t stream) {
  (void)in_sizes; (void)n_in; (void)out_size; (void)ws_size;
  const float* meas  = (const float*)d_in[0];
  const float* basis = (const float*)d_in[1];
  const float* rho   = (const float*)d_in[2];
  const float* h0    = (const float*)d_in[3];
  const float* c0    = (const float*)d_in[4];
  const float* cWih  = (const float*)d_in[5];
  const float* cWhh  = (const float*)d_in[6];
  const float* cB    = (const float*)d_in[7];
  const float* pW    = (const float*)d_in[8];
  const float* pB    = (const float*)d_in[9];
  const float* w0ih  = (const float*)d_in[10];
  const float* w0hh  = (const float*)d_in[11];
  const float* b0_   = (const float*)d_in[12];
  const float* w0hr  = (const float*)d_in[13];
  const float* w1ih  = (const float*)d_in[14];
  const float* w1hh  = (const float*)d_in[15];
  const float* b1_   = (const float*)d_in[16];
  const float* w1hr  = (const float*)d_in[17];
  float* outF = (float*)d_out;

  char* ws = (char*)d_ws;
  size_t off = 0;
  auto take = [&](size_t bytes) -> void* {
    void* p = ws + off;
    off += (bytes + 255) & ~(size_t)255;
    return p;
  };
  __bf16* fcWih = (__bf16*)take((size_t)64 * 512 * 2);        // K=17->32 pad
  __bf16* fcWhh = (__bf16*)take((size_t)8 * 64 * 512 * 2);    // K=256
  __bf16* f0Wih = (__bf16*)take((size_t)64 * 512 * 2);
  __bf16* f0Whh = (__bf16*)take((size_t)64 * 512 * 2);
  __bf16* f0Whr = (__bf16*)take((size_t)8 * 2 * 512 * 2);
  __bf16* f1Wih = (__bf16*)take((size_t)64 * 512 * 2);
  __bf16* f1Whh = (__bf16*)take((size_t)64 * 512 * 2);
  __bf16* f1Whr = (__bf16*)take((size_t)8 * 2 * 512 * 2);
  __bf16* seqF  = (__bf16*)take((size_t)NBTILE * TSTEPS * 512 * 2);  // 8 MB
  __bf16* h1F   = (__bf16*)take((size_t)NBTILE * TSTEPS * 512 * 2);  // 8 MB

  auto nb = [](int total) { return dim3((total + 255) / 256); };
  fill_bfrag<<<nb(8 * 64 * 512), 256, 0, stream>>>(cWhh, fcWhh, 64, 8, 256);
  fill_bfrag<<<nb(1 * 64 * 512), 256, 0, stream>>>(cWih, fcWih, 64, 1, 17);
  fill_bfrag<<<nb(1 * 64 * 512), 256, 0, stream>>>(w0ih, f0Wih, 64, 1, 17);
  fill_bfrag<<<nb(1 * 64 * 512), 256, 0, stream>>>(w0hh, f0Whh, 64, 1, 32);
  fill_bfrag<<<nb(8 * 2 * 512),  256, 0, stream>>>(w0hr, f0Whr, 2, 8, 256);
  fill_bfrag<<<nb(1 * 64 * 512), 256, 0, stream>>>(w1ih, f1Wih, 64, 1, 32);
  fill_bfrag<<<nb(1 * 64 * 512), 256, 0, stream>>>(w1hh, f1Whh, 64, 1, 32);
  fill_bfrag<<<nb(8 * 2 * 512),  256, 0, stream>>>(w1hr, f1Whr, 2, 8, 256);

  qst_phaseA<<<NBTILE, 256, 0, stream>>>(meas, basis, rho, h0, c0, cB, pW, pB,
                                         fcWih, fcWhh, seqF, outF);
  qst_phaseB<<<NBTILE, 256, 0, stream>>>(seqF, f0Wih, f0Whh, f0Whr, b0_, h1F, nullptr);
  qst_phaseB<<<NBTILE, 256, 0, stream>>>(h1F,  f1Wih, f1Whh, f1Whr, b1_, nullptr, outF);
}
